// GraphAttention_8237747273795
// MI455X (gfx1250) — compile-verified, matrix operand order checked by reading, not run on hardware
//
#include <hip/hip_runtime.h>
#include <math.h>

// Problem constants (from reference): Z=2, N=2048, Ki=32, Dm=256, H=4, Dk=64
#define ZB   2
#define NN   2048
#define KI   32
#define DM   256
#define NH   4
#define DK   64
#define ROWS (ZB * NN)          // 4096 token rows

typedef __attribute__((ext_vector_type(16))) __bf16 v16bf;
typedef __attribute__((ext_vector_type(8)))  float  v8f;

// ---------------------------------------------------------------------------
// WMMA fragment helpers (CDNA5 wave32 16x16x32 bf16 layout)
// A fragment (16x32, M x K): lane L holds row m = L&15; elements 0..7 are the
// 8 contiguous K halves starting at 8*(L>>4); elements 8..15 start at
// 16 + 8*(L>>4).  Same pattern for B (K x 16) indexed by output column.
// Both map to two 16-byte (b128) loads from a row-major bf16 matrix.
// ---------------------------------------------------------------------------
__device__ __forceinline__ v16bf load_frag(const __bf16* rowBase, int laneHi) {
    union { v16bf v; uint4 u[2]; } r;
    const __bf16* p = rowBase + 8 * laneHi;
    r.u[0] = *(const uint4*)(p);
    r.u[1] = *(const uint4*)(p + 16);
    return r.v;
}

// D = A(16x256) * B(256x16) accumulated in f32 over 8 k-steps.
__device__ __forceinline__ v8f mm_tile(const __bf16* Arow, const __bf16* Brow,
                                       int laneHi, int ksteps) {
    v8f acc = {0.f, 0.f, 0.f, 0.f, 0.f, 0.f, 0.f, 0.f};
    for (int kt = 0; kt < ksteps; ++kt) {
        v16bf a = load_frag(Arow + kt * 32, laneHi);
        v16bf b = load_frag(Brow + kt * 32, laneHi);
        acc = __builtin_amdgcn_wmma_f32_16x16x32_bf16(
            false, a, false, b, (short)0, acc, false, false);
    }
    return acc;
}

// Dual-accumulator variant sharing the A fragments (gamma & beta columns).
__device__ __forceinline__ void mm_tile2(const __bf16* Arow,
                                         const __bf16* Bg, const __bf16* Bb,
                                         int laneHi, v8f& accG, v8f& accB) {
    for (int kt = 0; kt < 8; ++kt) {
        v16bf a  = load_frag(Arow + kt * 32, laneHi);
        v16bf bg = load_frag(Bg + kt * 32, laneHi);
        v16bf bb = load_frag(Bb + kt * 32, laneHi);
        accG = __builtin_amdgcn_wmma_f32_16x16x32_bf16(
            false, a, false, bg, (short)0, accG, false, false);
        accB = __builtin_amdgcn_wmma_f32_16x16x32_bf16(
            false, a, false, bb, (short)0, accB, false, false);
    }
}

// ---------------------------------------------------------------------------
// Prep kernels: fp32 -> bf16 (with layout transforms for B-operand matrices)
// ---------------------------------------------------------------------------
__global__ void cvt_kernel(const float* __restrict__ src, __bf16* __restrict__ dst, int n) {
    int i = blockIdx.x * 256 + threadIdx.x;
    if (i < n) dst[i] = (__bf16)src[i];
}

// Vectorized x4 convert for the big edges tensor (b128 load -> b64 store).
__global__ void cvt4_kernel(const float* __restrict__ src, __bf16* __restrict__ dst, int n4) {
    int i = blockIdx.x * 256 + threadIdx.x;
    if (i < n4) {
        float4 v = ((const float4*)src)[i];
        union { __bf16 h[4]; uint2 u; } o;
        o.h[0] = (__bf16)v.x; o.h[1] = (__bf16)v.y;
        o.h[2] = (__bf16)v.z; o.h[3] = (__bf16)v.w;
        ((uint2*)dst)[i] = o.u;
    }
}

// dst[o][i] = src[i][o]  (src is [IN][OUT] row-major, dst is [OUT][IN])
__global__ void cvtT_kernel(const float* __restrict__ src, __bf16* __restrict__ dst,
                            int OUT, int IN) {
    int idx = blockIdx.x * 256 + threadIdx.x;
    if (idx < OUT * IN) {
        int o = idx / IN, i = idx % IN;
        dst[idx] = (__bf16)src[(size_t)i * OUT + o];
    }
}

// proj [H][Dm][Dk] -> w[c][d] with c = h*64+e : w[c][d] = proj[h][d][e]
__global__ void cvtProj_kernel(const float* __restrict__ src, __bf16* __restrict__ dst) {
    int idx = blockIdx.x * 256 + threadIdx.x;   // idx = c*256 + d
    if (idx < DM * DM) {
        int c = idx >> 8, d = idx & 255;
        int h = c >> 6, e = c & 63;
        dst[idx] = (__bf16)src[((size_t)h * DM + d) * DK + e];
    }
}

// ---------------------------------------------------------------------------
// Generic WMMA GEMM: C[m][c] = sum_d A[m][d]*W[c][d] (+bias[c]); fp32 out.
// Block = 128 threads = 4 waves; each block does a 16(M) x 256(N) tile.
// ---------------------------------------------------------------------------
__global__ __launch_bounds__(128) void gemm_bf16_kernel(
    const __bf16* __restrict__ A, const __bf16* __restrict__ W,
    const float* __restrict__ bias, float* __restrict__ C) {
    int wave = threadIdx.x >> 5;
    int lane = threadIdx.x & 31;
    int laneHi = lane >> 4, lane15 = lane & 15;
    int m = blockIdx.x * 16 + lane15;
    for (int j = 0; j < 4; ++j) {
        int c = (wave * 4 + j) * 16 + lane15;
        __builtin_prefetch(W + (size_t)c * DM, 0, 1);   // global_prefetch_b8
        v8f acc = mm_tile(A + (size_t)m * DM, W + (size_t)c * DM, laneHi, 8);
        float bb = bias ? bias[c] : 0.f;
        for (int r = 0; r < 8; ++r) {
            int mm = blockIdx.x * 16 + r + 8 * laneHi;
            C[(size_t)mm * DM + c] = acc[r] + bb;
        }
    }
}

// ---------------------------------------------------------------------------
// FiLM chain (device): hbuf = silu(sE @ W1 + b1);
//                      xbuf = (hbuf @ W2 + b2)[:, :256] * nj + (...)[:,256:]
// ---------------------------------------------------------------------------
__device__ void film_chain(const __bf16* sE, __bf16* hbuf, __bf16* xbuf,
                           const __bf16* W1, const __bf16* W2,
                           const float* b1, const float* b2,
                           const float* njRow,
                           int wave, int laneHi, int lane15) {
    // layer 1: silu
    for (int mt = 0; mt < 2; ++mt)
        for (int j = 0; j < 4; ++j) {
            int c = (wave * 4 + j) * 16 + lane15;
            int m = mt * 16 + lane15;
            v8f acc = mm_tile(sE + m * DM, W1 + (size_t)c * DM, laneHi, 8);
            float bb = b1[c];
            for (int r = 0; r < 8; ++r) {
                float x = acc[r] + bb;
                float s = x / (1.f + __expf(-x));      // silu
                hbuf[(mt * 16 + r + 8 * laneHi) * DM + c] = (__bf16)s;
            }
        }
    __syncthreads();
    // layer 2 (gamma & beta share A fragments) + FiLM combine with nodes_j
    for (int mt = 0; mt < 2; ++mt)
        for (int j = 0; j < 4; ++j) {
            int cg = (wave * 4 + j) * 16 + lane15;
            int m = mt * 16 + lane15;
            v8f aG = {0.f, 0.f, 0.f, 0.f, 0.f, 0.f, 0.f, 0.f};
            v8f aB = {0.f, 0.f, 0.f, 0.f, 0.f, 0.f, 0.f, 0.f};
            mm_tile2(hbuf + m * DM, W2 + (size_t)cg * DM,
                     W2 + (size_t)(256 + cg) * DM, laneHi, aG, aB);
            float bg = b2[cg], bt = b2[256 + cg];
            for (int r = 0; r < 8; ++r) {
                int mm = mt * 16 + r + 8 * laneHi;
                float g  = aG[r] + bg;
                float be = aB[r] + bt;
                float nj = njRow[(size_t)mm * DM + cg];
                xbuf[mm * DM + cg] = (__bf16)(g * nj + be);
            }
        }
    __syncthreads();
}

// proj: dst[m][c] = sum_d src[m][d]*W[c][d] + bias[c]   (M=32, N=256, K=256)
__device__ void proj256(const __bf16* sIn, __bf16* sOut, const __bf16* W,
                        const float* bias, int wave, int laneHi, int lane15) {
    for (int mt = 0; mt < 2; ++mt)
        for (int j = 0; j < 4; ++j) {
            int c = (wave * 4 + j) * 16 + lane15;
            int m = mt * 16 + lane15;
            v8f acc = mm_tile(sIn + m * DM, W + (size_t)c * DM, laneHi, 8);
            float bb = bias[c];
            for (int r = 0; r < 8; ++r)
                sOut[(mt * 16 + r + 8 * laneHi) * DM + c] = (__bf16)(acc[r] + bb);
        }
    __syncthreads();
}

// ---------------------------------------------------------------------------
// Attention kernel: one workgroup per (z,n) row. 128 threads = 4 waves.
// Edge tile arrives via CDNA5 async-to-LDS DMA (ASYNCcnt-tracked).
// ---------------------------------------------------------------------------
__global__ __launch_bounds__(128) void attn_kernel(
    const __bf16* __restrict__ edgesB, const float* __restrict__ nodes_j,
    const int* __restrict__ nbr_mask,
    const float* __restrict__ fk_b1, const float* __restrict__ fk_b2,
    const float* __restrict__ fv_b1, const float* __restrict__ fv_b2,
    const float* __restrict__ k_bias, const float* __restrict__ v_bias,
    const __bf16* __restrict__ wfk1, const __bf16* __restrict__ wfk2,
    const __bf16* __restrict__ wfv1, const __bf16* __restrict__ wfv2,
    const __bf16* __restrict__ wk, const __bf16* __restrict__ wv,
    const float* __restrict__ Qws, __bf16* __restrict__ attOut) {
    __shared__ __bf16 sE[KI * DM];   // edges (bf16), live whole kernel
    __shared__ __bf16 sB1[KI * DM];  // h_K -> K -> x_V
    __shared__ __bf16 sB2[KI * DM];  // x_K -> h_V -> V
    __shared__ float  sP[NH * KI];   // softmax probs

    const int row = blockIdx.x;                  // z*N + n
    const size_t tbase = (size_t)row * KI * DM;  // base into [Z,N,Ki,Dm]
    const int wave = threadIdx.x >> 5;
    const int lane = threadIdx.x & 31;
    const int laneHi = lane >> 4, lane15 = lane & 15;
    const float* njRow = nodes_j + tbase;

    // --- async DMA: 16KB bf16 edge tile -> LDS (8 x b128 per thread) ---
    {
        // generic shared pointer's low 32 bits are the LDS byte offset
        unsigned int ldsBase = (unsigned int)(uintptr_t)&sE[0] + threadIdx.x * 16u;
        unsigned long long ga =
            (unsigned long long)(uintptr_t)(edgesB + tbase) + threadIdx.x * 16ull;
        for (int r2 = 0; r2 < 8; ++r2) {
            unsigned int lo = ldsBase + (unsigned int)(r2 * 2048);
            unsigned long long g = ga + (unsigned long long)(r2 * 2048);
            asm volatile("global_load_async_to_lds_b128 %0, %1, off"
                         :: "v"(lo), "v"(g) : "memory");
        }
        asm volatile("s_wait_asynccnt 0" ::: "memory");
    }
    __syncthreads();

    // ---- FiLM-K chain: sB1 = h, sB2 = x_in ----
    film_chain(sE, sB1, sB2, wfk1, wfk2, fk_b1, fk_b2, njRow, wave, laneHi, lane15);
    // ---- K projection: sB1 = K (bf16) ----
    proj256(sB2, sB1, wk, k_bias, wave, laneHi, lane15);

    // ---- scores + softmax: wave = head, lane = neighbor ----
    {
        const int h = wave, k = lane;
        const float* q = Qws + (size_t)row * DM + h * DK;
        const __bf16* kr = sB1 + k * DM + h * DK;
        float s = 0.f;
        for (int e = 0; e < DK; ++e) s += q[e] * (float)kr[e];
        s *= 0.125f;                              // 1/sqrt(64)
        int inv = (nbr_mask[(size_t)row * KI + k] == 0) ? 1 : 0;
        int allinv = inv;
        for (int off = 16; off > 0; off >>= 1) allinv &= __shfl_xor(allinv, off, 32);
        if (inv && !allinv) s = -INFINITY;
        float mx = s;
        for (int off = 16; off > 0; off >>= 1) mx = fmaxf(mx, __shfl_xor(mx, off, 32));
        float ex = __expf(s - mx);
        float sum = ex;
        for (int off = 16; off > 0; off >>= 1) sum += __shfl_xor(sum, off, 32);
        sP[h * KI + k] = ex / sum;
    }
    __syncthreads();

    // ---- FiLM-V chain (reuse buffers): sB2 = h2, sB1 = x2 ----
    film_chain(sE, sB2, sB1, wfv1, wfv2, fv_b1, fv_b2, njRow, wave, laneHi, lane15);
    // ---- V projection: sB2 = V (bf16) ----
    proj256(sB1, sB2, wv, v_bias, wave, laneHi, lane15);

    // ---- out[h][e] = sum_k P[h,k] * V[k, h*64+e]; store in (e,H) order ----
    for (int c = threadIdx.x; c < DM; c += 128) {
        const int h = c >> 6, e = c & 63;
        const float* p = sP + h * KI;
        float o = 0.f;
        for (int k = 0; k < KI; ++k) o += p[k] * (float)sB2[k * DM + c];
        attOut[(size_t)row * DM + (e * NH + h)] = (__bf16)o;
    }
}

// ---------------------------------------------------------------------------
// Host-side launch
// ---------------------------------------------------------------------------
extern "C" void kernel_launch(void* const* d_in, const int* in_sizes, int n_in,
                              void* d_out, int out_size, void* d_ws, size_t ws_size,
                              hipStream_t stream) {
    (void)in_sizes; (void)n_in; (void)out_size; (void)ws_size;
    const float* nodes_i = (const float*)d_in[0];
    const float* edges   = (const float*)d_in[1];
    const float* nodes_j = (const float*)d_in[2];
    const int*   nbrm    = (const int*)d_in[3];
    const float* fk_W1 = (const float*)d_in[4];
    const float* fk_b1 = (const float*)d_in[5];
    const float* fk_W2 = (const float*)d_in[6];
    const float* fk_b2 = (const float*)d_in[7];
    const float* fv_W1 = (const float*)d_in[8];
    const float* fv_b1 = (const float*)d_in[9];
    const float* fv_W2 = (const float*)d_in[10];
    const float* fv_b2 = (const float*)d_in[11];
    const float* q_proj = (const float*)d_in[12];
    const float* k_proj = (const float*)d_in[13];
    const float* v_proj = (const float*)d_in[14];
    const float* q_bias = (const float*)d_in[15];
    const float* k_bias = (const float*)d_in[16];
    const float* v_bias = (const float*)d_in[17];
    const float* out_W  = (const float*)d_in[18];

    // workspace carve-up
    char* ws = (char*)d_ws;
    size_t off = 0;
    auto take = [&](size_t bytes) { char* p = ws + off; off += (bytes + 255) & ~size_t(255); return p; };
    __bf16* edgesB = (__bf16*)take((size_t)ROWS * KI * DM * 2);   // 64 MiB
    __bf16* niB   = (__bf16*)take((size_t)ROWS * DM * 2);
    __bf16* wfk1  = (__bf16*)take((size_t)DM * DM * 2);
    __bf16* wfk2  = (__bf16*)take((size_t)2 * DM * DM * 2);
    __bf16* wfv1  = (__bf16*)take((size_t)DM * DM * 2);
    __bf16* wfv2  = (__bf16*)take((size_t)2 * DM * DM * 2);
    __bf16* wq    = (__bf16*)take((size_t)DM * DM * 2);
    __bf16* wk    = (__bf16*)take((size_t)DM * DM * 2);
    __bf16* wv    = (__bf16*)take((size_t)DM * DM * 2);
    __bf16* wout  = (__bf16*)take((size_t)DM * DM * 2);
    float*  Qws   = (float*)take((size_t)ROWS * DM * 4);
    __bf16* attO  = (__bf16*)take((size_t)ROWS * DM * 2);

    // --- prep: bf16 conversion + transposes ---
    const int nEdge4 = ROWS * KI * DM / 4;
    cvt4_kernel<<<(nEdge4 + 255) / 256, 256, 0, stream>>>(edges, edgesB, nEdge4);
    cvt_kernel<<<(ROWS * DM + 255) / 256, 256, 0, stream>>>(nodes_i, niB, ROWS * DM);
    cvt_kernel<<<(DM * DM + 255) / 256, 256, 0, stream>>>(out_W, wout, DM * DM);
    cvtT_kernel<<<(DM * DM + 255) / 256, 256, 0, stream>>>(fk_W1, wfk1, DM, DM);
    cvtT_kernel<<<(2 * DM * DM + 255) / 256, 256, 0, stream>>>(fk_W2, wfk2, 2 * DM, DM);
    cvtT_kernel<<<(DM * DM + 255) / 256, 256, 0, stream>>>(fv_W1, wfv1, DM, DM);
    cvtT_kernel<<<(2 * DM * DM + 255) / 256, 256, 0, stream>>>(fv_W2, wfv2, 2 * DM, DM);
    cvtProj_kernel<<<(DM * DM + 255) / 256, 256, 0, stream>>>(q_proj, wq);
    cvtProj_kernel<<<(DM * DM + 255) / 256, 256, 0, stream>>>(k_proj, wk);
    cvtProj_kernel<<<(DM * DM + 255) / 256, 256, 0, stream>>>(v_proj, wv);

    // --- Q = nodes_i @ wq^T + q_bias  (WMMA GEMM, fp32 out) ---
    gemm_bf16_kernel<<<ROWS / 16, 128, 0, stream>>>(niB, wq, q_bias, Qws);

    // --- fused FiLM + KV projection + attention per (z,n) ---
    attn_kernel<<<ROWS, 128, 0, stream>>>(edgesB, nodes_j, nbrm,
                                          fk_b1, fk_b2, fv_b1, fv_b2,
                                          k_bias, v_bias,
                                          wfk1, wfk2, wfv1, wfv2, wk, wv,
                                          Qws, attO);

    // --- final projection: out = attO @ out_W^T (WMMA GEMM, fp32 out) ---
    gemm_bf16_kernel<<<ROWS / 16, 128, 0, stream>>>(attO, wout, nullptr, (float*)d_out);
}
